// GNN_79517024518974
// MI455X (gfx1250) — matrix operand order repaired and checked by reading, not compiled, not run on hardware
//
#include <hip/hip_runtime.h>
#include <hip/hip_bf16.h>

typedef __attribute__((ext_vector_type(16))) __bf16 v16bf;
typedef __attribute__((ext_vector_type(8)))  float  v8f;

#define NEG_SLOPE 0.2f

// ---- float atomic max via monotonic int/uint trick (always available) ----
__device__ inline void atomicMaxF32(float* addr, float val) {
    if (val >= 0.0f) atomicMax((int*)addr, __float_as_int(val));
    else             atomicMin((unsigned int*)addr, __float_as_uint(val));
}

// =====================================================================
// GEMM: H[N x DOUT] = X[N x 128] * W[128 x DOUT], fp32 in/out, bf16 WMMA.
// Block = 256 threads = 8 waves; wave w owns rows [blk*128 + w*16, +16).
// W is staged transposed (W^T[n][k]) in LDS as bf16.
// =====================================================================
template<int DOUT>
__global__ __launch_bounds__(256)
void gat_gemm(const float* __restrict__ X, const float* __restrict__ W,
              float* __restrict__ H, int N)
{
    __shared__ __attribute__((aligned(32))) __bf16 Wlds[128 * DOUT]; // [n][k]
    const int tid = threadIdx.x;
    for (int i = tid; i < 128 * DOUT; i += 256) {
        const int n = i >> 7, k = i & 127;
        Wlds[i] = (__bf16)W[k * DOUT + n];
    }
    __syncthreads();

    const int wave  = tid >> 5;
    const int lane  = tid & 31;
    const int half  = lane >> 4;   // 0: lanes 0-15, 1: lanes 16-31
    const int l16   = lane & 15;
    const int tileM = blockIdx.x * 128 + wave * 16;
    const int row   = tileM + l16;           // A-matrix row for this lane

    v8f acc[DOUT / 16];
#pragma unroll
    for (int t = 0; t < DOUT / 16; ++t)
#pragma unroll
        for (int r = 0; r < 8; ++r) acc[t][r] = 0.0f;

#pragma unroll
    for (int k0 = 0; k0 < 128; k0 += 32) {
        // A fragment, 16-bit A 16x32 layout (contiguous K, matches B):
        // lane<16 : elems 0..15 = K k0..k0+15
        // lane>=16: elems 0..15 = K k0+16..k0+31
        float lo[8], hi[8];
        if (row < N) {
            const float* p = X + (size_t)row * 128 + k0 + half * 16;
#pragma unroll
            for (int j = 0; j < 8; ++j) { lo[j] = p[j]; hi[j] = p[j + 8]; }
        } else {
#pragma unroll
            for (int j = 0; j < 8; ++j) { lo[j] = 0.0f; hi[j] = 0.0f; }
        }
        v16bf a;
#pragma unroll
        for (int j = 0; j < 8; ++j) {
            a[j]     = (__bf16)lo[j];
            a[j + 8] = (__bf16)hi[j];
        }
#pragma unroll
        for (int t = 0; t < DOUT / 16; ++t) {
            // B fragment, 16-bit B 32x16: col = lane&15;
            // lane<16: elems e = K k0+e ; lane>=16: elems e = K k0+16+e
            const v16bf b = *reinterpret_cast<const v16bf*>(
                &Wlds[(t * 16 + l16) * 128 + k0 + half * 16]);
            acc[t] = __builtin_amdgcn_wmma_f32_16x16x32_bf16(
                false, a, false, b, (short)0, acc[t], false, false);
        }
    }

    // C/D layout: VGPR r -> row tileM + half*8 + r, col l16
#pragma unroll
    for (int t = 0; t < DOUT / 16; ++t) {
#pragma unroll
        for (int r = 0; r < 8; ++r) {
            const int m = tileM + half * 8 + r;
            if (m < N) H[(size_t)m * DOUT + t * 16 + l16] = acc[t][r];
        }
    }
}

// ---- per-node attention scores: s_src[n]=h[n]·a_src, s_dst[n]=h[n]·a_dst ----
__global__ void attn_scores(const float* __restrict__ H,
                            const float* __restrict__ a_src,
                            const float* __restrict__ a_dst,
                            float* __restrict__ ssrc, float* __restrict__ sdst,
                            int N, int dout)
{
    const int w    = (blockIdx.x * blockDim.x + threadIdx.x) >> 5;
    const int lane = threadIdx.x & 31;
    if (w >= N) return;
    float ss = 0.0f, sd = 0.0f;
    for (int k = lane; k < dout; k += 32) {
        const float hv = H[(size_t)w * dout + k];
        ss += hv * a_src[k];
        sd += hv * a_dst[k];
    }
    for (int off = 16; off > 0; off >>= 1) {
        ss += __shfl_xor(ss, off, 32);
        sd += __shfl_xor(sd, off, 32);
    }
    if (lane == 0) { ssrc[w] = ss; sdst[w] = sd; }
}

__global__ void init_md(float* __restrict__ mx, float* __restrict__ den, int N)
{
    const int i = blockIdx.x * blockDim.x + threadIdx.x;
    if (i < N) { mx[i] = -3.0e38f; den[i] = 0.0f; }
}

__global__ void zero_f32(float* __restrict__ p, int n)
{
    const int i = blockIdx.x * blockDim.x + threadIdx.x;
    if (i < n) p[i] = 0.0f;
}

// ---- edge pass 1: e = leaky_relu(s_src[src]+s_dst[dst]); segment max ----
__global__ void edge_score_max(const int* __restrict__ src, const int* __restrict__ dst,
                               const float* __restrict__ ssrc, const float* __restrict__ sdst,
                               float* __restrict__ ebuf, float* __restrict__ mx,
                               int E, int Etot)
{
    const int i = blockIdx.x * blockDim.x + threadIdx.x;
    if (i >= Etot) return;
    int s, d;
    if (i < E) { s = src[i]; d = dst[i]; } else { s = d = i - E; } // self-loops
    float e = ssrc[s] + sdst[d];
    e = (e > 0.0f) ? e : NEG_SLOPE * e;
    ebuf[i] = e;
    atomicMaxF32(&mx[d], e);
}

// ---- edge pass 2: a = exp(e - m[dst]); segment sum into denom ----
__global__ void edge_exp_sum(const int* __restrict__ dst,
                             const float* __restrict__ mx,
                             float* __restrict__ ebuf, float* __restrict__ den,
                             int E, int Etot)
{
    const int i = blockIdx.x * blockDim.x + threadIdx.x;
    if (i >= Etot) return;
    const int d = (i < E) ? dst[i] : (i - E);
    const float a = __expf(ebuf[i] - mx[d]);
    ebuf[i] = a;
    atomicAdd(&den[d], a);
}

// ---- edge pass 3: agg[dst] += a * h[src]; one wave32 per edge ----
__global__ void edge_aggregate(const int* __restrict__ src, const int* __restrict__ dst,
                               const float* __restrict__ att, const float* __restrict__ H,
                               float* __restrict__ agg, int E, int Etot, int dout)
{
    const int w    = (blockIdx.x * blockDim.x + threadIdx.x) >> 5;
    const int lane = threadIdx.x & 31;
    if (w >= Etot) return;
    int s, d;
    if (w < E) { s = src[w]; d = dst[w]; } else { s = d = w - E; }
    const float a = att[w];
    for (int k = lane; k < dout; k += 32)
        atomicAdd(&agg[(size_t)d * dout + k], H[(size_t)s * dout + k] * a);
}

// ---- x = agg/denom + bias (fused layer-output -> next-layer-input) ----
__global__ void normalize_bias(float* __restrict__ x, const float* __restrict__ den,
                               const float* __restrict__ b, int N, int dout)
{
    const int i = blockIdx.x * blockDim.x + threadIdx.x;
    if (i >= N * dout) return;
    const int n = i / dout, k = i % dout;
    x[i] = x[i] / den[n] + b[k];
}

// ---- final: out[k] = mean_n (agg[n,k]/den[n] + b[k]), dout = 64 ----
__global__ void mean_out(const float* __restrict__ agg, const float* __restrict__ den,
                         const float* __restrict__ b, float* __restrict__ out,
                         int N, float invN)
{
    __shared__ float sm[256];
    const int t = threadIdx.x;
    const int k = t & 63, g = t >> 6;
    const int n0 = blockIdx.x * 512;
    float local = 0.0f;
    for (int n = n0 + g; n < n0 + 512 && n < N; n += 4)
        local += agg[(size_t)n * 64 + k] / den[n] + b[k];
    sm[t] = local;
    __syncthreads();
    if (g == 0) {
        const float tot = sm[t] + sm[t + 64] + sm[t + 128] + sm[t + 192];
        atomicAdd(&out[k], tot * invN);
    }
}

extern "C" void kernel_launch(void* const* d_in, const int* in_sizes, int n_in,
                              void* d_out, int out_size, void* d_ws, size_t ws_size,
                              hipStream_t stream)
{
    const float* x   = (const float*)d_in[0];
    const int*   ei  = (const int*)d_in[1];
    const float* W1  = (const float*)d_in[2];
    const float* as1 = (const float*)d_in[3];
    const float* ad1 = (const float*)d_in[4];
    const float* b1  = (const float*)d_in[5];
    const float* W2  = (const float*)d_in[6];
    const float* as2 = (const float*)d_in[7];
    const float* ad2 = (const float*)d_in[8];
    const float* b2  = (const float*)d_in[9];
    const float* W3  = (const float*)d_in[10];
    const float* as3 = (const float*)d_in[11];
    const float* ad3 = (const float*)d_in[12];
    const float* b3  = (const float*)d_in[13];

    const int N    = in_sizes[0] / 128;
    const int E    = in_sizes[1] / 2;
    const int Etot = E + N;          // + self-loops
    const int* srcI = ei;
    const int* dstI = ei + E;

    float* ws   = (float*)d_ws;
    float* h    = ws;                         // N * 128
    float* agg  = h   + (size_t)N * 128;      // N * 128
    float* ssrc = agg + (size_t)N * 128;      // N
    float* sdst = ssrc + N;                   // N
    float* mx   = sdst + N;                   // N
    float* den  = mx   + N;                   // N
    float* ebuf = den  + N;                   // Etot
    float* out  = (float*)d_out;              // 64 fp32

    auto cdiv = [](int a, int b) { return (a + b - 1) / b; };
    const int gemmBlocks     = cdiv(N, 128);
    const int nodeWaveBlocks = cdiv(N * 32, 256);
    const int nodeBlocks     = cdiv(N, 256);
    const int edgeBlocks     = cdiv(Etot, 256);
    const int edgeWaveBlocks = cdiv(Etot * 32, 256);

    auto run_layer = [&](const float* in, const float* W, const float* as,
                         const float* ad, int dout) {
        if (dout == 128)
            gat_gemm<128><<<gemmBlocks, 256, 0, stream>>>(in, W, h, N);
        else
            gat_gemm<64><<<gemmBlocks, 256, 0, stream>>>(in, W, h, N);
        attn_scores<<<nodeWaveBlocks, 256, 0, stream>>>(h, as, ad, ssrc, sdst, N, dout);
        init_md<<<nodeBlocks, 256, 0, stream>>>(mx, den, N);
        zero_f32<<<cdiv(N * dout, 256), 256, 0, stream>>>(agg, N * dout);
        edge_score_max<<<edgeBlocks, 256, 0, stream>>>(srcI, dstI, ssrc, sdst, ebuf, mx, E, Etot);
        edge_exp_sum<<<edgeBlocks, 256, 0, stream>>>(dstI, mx, ebuf, den, E, Etot);
        edge_aggregate<<<edgeWaveBlocks, 256, 0, stream>>>(srcI, dstI, ebuf, h, agg, E, Etot, dout);
    };

    // layer 1: input = x
    run_layer(x, W1, as1, ad1, 128);
    // layer 2: input = agg1/den1 + b1 (normalized in place)
    normalize_bias<<<cdiv(N * 128, 256), 256, 0, stream>>>(agg, den, b1, N, 128);
    run_layer(agg, W2, as2, ad2, 128);
    // layer 3: input = agg2/den2 + b2
    normalize_bias<<<cdiv(N * 128, 256), 256, 0, stream>>>(agg, den, b2, N, 128);
    run_layer(agg, W3, as3, ad3, 64);
    // final mean over nodes (normalize by den3 + b3 fused)
    zero_f32<<<1, 64, 0, stream>>>(out, 64);
    mean_out<<<cdiv(N, 512), 256, 0, stream>>>(agg, den, b3, out, N, 1.0f / (float)N);
}
